// MultiHeadAttention_67319317397622
// MI455X (gfx1250) — matrix operand order repaired and checked by reading, not compiled, not run on hardware
//
#include <hip/hip_runtime.h>

// ---------------------------------------------------------------------------
// MHA forward for MI455X (gfx1250): bf16 WMMA (16x16x32) with f32 accumulate.
// B=2, N=2048, C=1024, H=16, D=64.
//  - double-buffered LDS staging, 1 barrier per K-step
//  - sched_barrier fence so each 8-WMMA burst sits behind ONE s_wait_dscnt
//  - GLOBAL_LOAD_ASYNC_TO_LDS_B128 (ASYNCcnt) for pure-bf16 staging copies
//  - global_prefetch 2 tiles ahead in the GEMMs
// ---------------------------------------------------------------------------

typedef __attribute__((ext_vector_type(16))) __bf16 v16bf;
typedef __attribute__((ext_vector_type(8)))  __bf16 v8bf;
typedef __attribute__((ext_vector_type(8)))  float  v8f;

union Frag16 { v16bf v; v8bf h[2]; };

#if defined(__has_builtin)
#if __has_builtin(__builtin_amdgcn_sched_barrier)
#define SCHED_FENCE() __builtin_amdgcn_sched_barrier(0)
#endif
#endif
#ifndef SCHED_FENCE
#define SCHED_FENCE()
#endif

// Async memory -> LDS copy (16 bytes per lane), tracked by ASYNCcnt.
__device__ static inline void async_copy_b128(const void* lds_dst, const void* gaddr) {
  const unsigned           ldsOff = (unsigned)(uintptr_t)lds_dst;       // low 32b = LDS offset
  const unsigned long long ga     = (unsigned long long)(uintptr_t)gaddr;
  asm volatile("global_load_async_to_lds_b128 %0, %1, off"
               :: "v"(ldsOff), "v"(ga) : "memory");
}
__device__ static inline void async_wait0() {
  asm volatile("s_wait_asynccnt 0x0" ::: "memory");
}

__device__ static inline v8f vzero8f() {
  v8f v;
  #pragma unroll
  for (int i = 0; i < 8; ++i) v[i] = 0.0f;
  return v;
}

__device__ static inline v8f wmma_bf16(v16bf a, v16bf b, v8f c) {
  // D = A(16x32 bf16) * B(32x16 bf16) + C(16x16 f32)
  return __builtin_amdgcn_wmma_f32_16x16x32_bf16(
      /*neg_a=*/false, a, /*neg_b=*/false, b,
      /*c_mod=*/(short)0, c, /*reuse_a=*/false, /*reuse_b=*/false);
}

// A-fragment (16x32, 16-bit): lane L<16 -> row L, K = {0..7, 16..23};
// lane L>=16 -> row L-16, K = {8..15, 24..31}.  base points at row 0, K 0.
__device__ static inline v16bf load_a_frag(const __bf16* base, int rowStride) {
  const int lane = threadIdx.x & 31;
  const int r  = lane & 15;
  const int k0 = (lane >> 4) << 3;         // 0 or 8
  const __bf16* p = base + r * rowStride + k0;
  Frag16 f;
  f.h[0] = *(const v8bf*)(p);              // K = k0 .. k0+7
  f.h[1] = *(const v8bf*)(p + 16);         // K = k0+16 .. k0+23
  return f.v;
}

// B-fragment (32x16, 16-bit): lane L -> col L%16, K = {0..7, 16..23} (L<16)
// or {8..15, 24..31} (L>=16), matching the A-fragment order.
__device__ static inline v16bf load_b_frag(const __bf16* base, int rowStride) {
  const int lane = threadIdx.x & 31;
  const int k0 = (lane >> 4) << 3;
  const __bf16* p = base + k0 * rowStride + (lane & 15);
  v16bf v;
  #pragma unroll
  for (int i = 0; i < 8; ++i) {
    v[i]     = p[i * rowStride];
    v[8 + i] = p[(16 + i) * rowStride];
  }
  return v;
}

// ---------------------------------------------------------------------------
// Kernel 1: QKV GEMM  (4096x1024)*(1024x3072) + bias, scatter to Q/Kt/V (bf16)
//   Q : [bh][n][d], pre-scaled by 1/sqrt(64)
//   Kt: [bh][d][n]  (transposed for the attention B-fragment)
//   V : [bh][n][d]
// fp32 inputs must be converted to bf16, so staging goes through registers.
// ---------------------------------------------------------------------------
__global__ __launch_bounds__(256)
void qkv_gemm_kernel(const float* __restrict__ x, const float* __restrict__ Wqkv,
                     const float* __restrict__ bqkv,
                     __bf16* __restrict__ Qs, __bf16* __restrict__ Kt,
                     __bf16* __restrict__ Vs) {
  constexpr int AS = 40;    // LDS row stride (bf16) for 128x32 A tile
  constexpr int BS = 136;   // LDS row stride (bf16) for 32x128 B tile
  __shared__ __bf16 As[2][128 * AS];
  __shared__ __bf16 Bs[2][32 * BS];

  const int t = threadIdx.x;
  const int lane = t & 31;
  const int wave = t >> 5;
  const int mBase = blockIdx.y * 128;
  const int nBase = blockIdx.x * 128;
  const int waveM = (wave & 3) * 32;
  const int waveN = (wave >> 2) * 64;

  float4 aReg[4], bReg[4];
  const int ac = (t & 7) * 4,  ar = t >> 3;   // A: col 0..28, row 0..31 (+32i)
  const int bc = (t & 31) * 4, br = t >> 5;   // B: col 0..124, row 0..7 (+8i)

  auto loadA = [&](int kk) {
    #pragma unroll
    for (int i = 0; i < 4; ++i)
      aReg[i] = *(const float4*)(x + (size_t)(mBase + ar + 32 * i) * 1024 + kk + ac);
  };
  auto loadB = [&](int kk) {
    #pragma unroll
    for (int i = 0; i < 4; ++i)
      bReg[i] = *(const float4*)(Wqkv + (size_t)(kk + br + 8 * i) * 3072 + nBase + bc);
  };
  auto storeTiles = [&](int buf) {
    #pragma unroll
    for (int i = 0; i < 4; ++i) {
      __bf16* da = &As[buf][(ar + 32 * i) * AS + ac];
      da[0] = (__bf16)aReg[i].x; da[1] = (__bf16)aReg[i].y;
      da[2] = (__bf16)aReg[i].z; da[3] = (__bf16)aReg[i].w;
      __bf16* db = &Bs[buf][(br + 8 * i) * BS + bc];
      db[0] = (__bf16)bReg[i].x; db[1] = (__bf16)bReg[i].y;
      db[2] = (__bf16)bReg[i].z; db[3] = (__bf16)bReg[i].w;
    }
  };

  v8f acc[2][4];
  #pragma unroll
  for (int i = 0; i < 2; ++i)
    #pragma unroll
    for (int j = 0; j < 4; ++j) acc[i][j] = vzero8f();

  loadA(0); loadB(0);
  storeTiles(0);
  __syncthreads();

  int cur = 0;
  for (int k0 = 0; k0 < 1024; k0 += 32) {
    const bool more = (k0 + 32) < 1024;
    if (more) { loadA(k0 + 32); loadB(k0 + 32); }
    if (k0 + 64 < 1024) {   // pull 2-ahead cachelines toward GL2
      __builtin_prefetch(x + (size_t)(mBase + ar) * 1024 + k0 + 64 + ac, 0, 1);
      __builtin_prefetch(Wqkv + (size_t)(k0 + 64 + br) * 3072 + nBase + bc, 0, 1);
    }

    // All 6 fragment loads issue before any WMMA (one dscnt wait per K-step).
    const v16bf a0 = load_a_frag(&As[cur][(waveM +  0) * AS], AS);
    const v16bf a1 = load_a_frag(&As[cur][(waveM + 16) * AS], AS);
    v16bf bf[4];
    #pragma unroll
    for (int nf = 0; nf < 4; ++nf)
      bf[nf] = load_b_frag(&Bs[cur][waveN + nf * 16], BS);
    SCHED_FENCE();
    #pragma unroll
    for (int nf = 0; nf < 4; ++nf) {
      acc[0][nf] = wmma_bf16(a0, bf[nf], acc[0][nf]);
      acc[1][nf] = wmma_bf16(a1, bf[nf], acc[1][nf]);
    }
    SCHED_FENCE();

    if (more) {
      storeTiles(cur ^ 1);
      __syncthreads();
      cur ^= 1;
    }
  }

  // Epilogue: add bias, de-interleave (h, d, s) packing, scatter bf16.
  const int colLane = lane & 15;
  const int rowSel = (lane >> 4) << 3;
  #pragma unroll
  for (int mf = 0; mf < 2; ++mf)
    #pragma unroll
    for (int nf = 0; nf < 4; ++nf)
      #pragma unroll
      for (int r = 0; r < 8; ++r) {
        const int row = mBase + waveM + mf * 16 + r + rowSel;   // b*2048 + n
        const int col = nBase + waveN + nf * 16 + colLane;      // 0..3071
        const float vc = acc[mf][nf][r] + bqkv[col];
        const int b = row >> 11, n = row & 2047;
        const int h = col / 192;
        const int rr = col - h * 192;
        const int d = rr / 3;
        const int s = rr - d * 3;
        const int bh = (b << 4) + h;
        if (s == 0)
          Qs[((size_t)bh * 2048 + n) * 64 + d] = (__bf16)(vc * 0.125f);
        else if (s == 1)
          Kt[((size_t)bh * 64 + d) * 2048 + n] = (__bf16)vc;
        else
          Vs[((size_t)bh * 2048 + n) * 64 + d] = (__bf16)vc;
      }
}

// ---------------------------------------------------------------------------
// Kernel 2: flash attention per (b,h).  K/V tiles staged with async DMA
// (GLOBAL_LOAD_ASYNC_TO_LDS_B128, ASYNCcnt) into a double buffer; Q frags
// resident in VGPRs; online softmax with 16-lane shuffles; P re-swizzled
// through per-wave LDS for P*V.  Output written into X2[b*2048+n][h*64+d].
// ---------------------------------------------------------------------------
__global__ __launch_bounds__(256)
void attn_kernel(const __bf16* __restrict__ Qs, const __bf16* __restrict__ Kt,
                 const __bf16* __restrict__ Vs, __bf16* __restrict__ X2) {
  constexpr int KS = 40;   // ktile row stride (bf16), rows = d (64), cols = n (32)
  constexpr int VS = 72;   // vtile row stride (bf16), rows = n (32), cols = d (64)
  constexpr int PS = 40;   // per-wave P staging stride, 32x32
  __shared__ __bf16 ktile[2][64 * KS];
  __shared__ __bf16 vtile[2][32 * VS];
  __shared__ __bf16 ptile[8][32 * PS];

  const int t = threadIdx.x;
  const int lane = t & 31;
  const int wave = t >> 5;
  const int bh = blockIdx.y;            // 0..31
  const int b = bh >> 4, h = bh & 15;
  const int qBase = blockIdx.x * 256 + wave * 32;

  const __bf16* Qbh  = Qs + (size_t)bh * 2048 * 64;
  const __bf16* Ktbh = Kt + (size_t)bh * 64 * 2048;
  const __bf16* Vbh  = Vs + (size_t)bh * 2048 * 64;

  // Async staging: each thread DMAs one 16B chunk of K^T and one of V.
  const int kd = t >> 2, kq = t & 3;    // ktile: d row (0..63), 8-elem quarter
  const int vn = t >> 3, vp = t & 7;    // vtile: n row (0..31), 8-elem part
  auto asyncKV = [&](int nb, int buf) {
    async_copy_b128(&ktile[buf][kd * KS + kq * 8],
                    Ktbh + (size_t)kd * 2048 + nb + kq * 8);
    async_copy_b128(&vtile[buf][vn * VS + vp * 8],
                    Vbh + (size_t)(nb + vn) * 64 + vp * 8);
  };

  // Q fragments: [mfrag][dchunk], resident for the whole kernel.
  v16bf qf[2][2];
  {
    const int r = lane & 15;
    const int k0 = (lane >> 4) << 3;
    #pragma unroll
    for (int mf = 0; mf < 2; ++mf)
      #pragma unroll
      for (int c = 0; c < 2; ++c) {
        const __bf16* p = Qbh + (size_t)(qBase + mf * 16 + r) * 64 + c * 32 + k0;
        Frag16 f;
        f.h[0] = *(const v8bf*)(p);
        f.h[1] = *(const v8bf*)(p + 16);
        qf[mf][c] = f.v;
      }
  }

  v8f o[2][4];
  float mrow[2][8], lrow[2][8];
  #pragma unroll
  for (int mf = 0; mf < 2; ++mf) {
    #pragma unroll
    for (int nf = 0; nf < 4; ++nf) o[mf][nf] = vzero8f();
    #pragma unroll
    for (int r = 0; r < 8; ++r) { mrow[mf][r] = -1e30f; lrow[mf][r] = 0.0f; }
  }

  asyncKV(0, 0);
  async_wait0();
  __syncthreads();

  int cur = 0;
  __bf16* pw = ptile[wave];
  for (int nb = 0; nb < 2048; nb += 32) {
    const bool more = (nb + 32) < 2048;
    if (more) asyncKV(nb + 32, cur ^ 1);   // DMA hides under the whole compute

    // S = Q * K^T  (scale folded into Q).  4 B-frags, fence, 8-WMMA burst.
    v16bf bk[2][2];
    #pragma unroll
    for (int c = 0; c < 2; ++c) {
      bk[c][0] = load_b_frag(&ktile[cur][(c * 32) * KS +  0], KS);
      bk[c][1] = load_b_frag(&ktile[cur][(c * 32) * KS + 16], KS);
    }
    v8f s[2][2];
    #pragma unroll
    for (int mf = 0; mf < 2; ++mf) { s[mf][0] = vzero8f(); s[mf][1] = vzero8f(); }
    SCHED_FENCE();
    #pragma unroll
    for (int c = 0; c < 2; ++c)
      #pragma unroll
      for (int mf = 0; mf < 2; ++mf) {
        s[mf][0] = wmma_bf16(qf[mf][c], bk[c][0], s[mf][0]);
        s[mf][1] = wmma_bf16(qf[mf][c], bk[c][1], s[mf][1]);
      }
    SCHED_FENCE();

    // Online softmax; stage P (bf16) to per-wave LDS in row-major 32x32.
    #pragma unroll
    for (int mf = 0; mf < 2; ++mf)
      #pragma unroll
      for (int r = 0; r < 8; ++r) {
        const float s0 = s[mf][0][r], s1 = s[mf][1][r];
        float mx = fmaxf(s0, s1);
        #pragma unroll
        for (int sh = 1; sh < 16; sh <<= 1) mx = fmaxf(mx, __shfl_xor(mx, sh, 32));
        const float mnew = fmaxf(mrow[mf][r], mx);
        const float corr = __expf(mrow[mf][r] - mnew);
        const float p0 = __expf(s0 - mnew);
        const float p1 = __expf(s1 - mnew);
        float rs = p0 + p1;
        #pragma unroll
        for (int sh = 1; sh < 16; sh <<= 1) rs += __shfl_xor(rs, sh, 32);
        lrow[mf][r] = lrow[mf][r] * corr + rs;
        mrow[mf][r] = mnew;
        #pragma unroll
        for (int nf = 0; nf < 4; ++nf) o[mf][nf][r] *= corr;
        const int prow = mf * 16 + r + ((lane >> 4) << 3);
        const int pcol = lane & 15;
        pw[prow * PS + pcol]      = (__bf16)p0;
        pw[prow * PS + 16 + pcol] = (__bf16)p1;
      }

    // O += P * V.  pa0/pa1 + 4 bv frags, fence, 8-WMMA burst.
    const v16bf pa0 = load_a_frag(&pw[0], PS);
    const v16bf pa1 = load_a_frag(&pw[16 * PS], PS);
    v16bf bv[4];
    #pragma unroll
    for (int nf = 0; nf < 4; ++nf)
      bv[nf] = load_b_frag(&vtile[cur][nf * 16], VS);
    SCHED_FENCE();
    #pragma unroll
    for (int nf = 0; nf < 4; ++nf) {
      o[0][nf] = wmma_bf16(pa0, bv[nf], o[0][nf]);
      o[1][nf] = wmma_bf16(pa1, bv[nf], o[1][nf]);
    }
    SCHED_FENCE();

    if (more) {
      async_wait0();       // own-wave DMA done; barrier makes it WG-visible
      __syncthreads();
      cur ^= 1;
    }
  }

  // Finalize and write into proj-input layout.
  const int colLane = lane & 15;
  const int rsel = (lane >> 4) << 3;
  #pragma unroll
  for (int mf = 0; mf < 2; ++mf)
    #pragma unroll
    for (int r = 0; r < 8; ++r) {
      const float inv = 1.0f / lrow[mf][r];
      const int n = qBase + mf * 16 + r + rsel;
      const size_t rowX = (size_t)b * 2048 + n;
      #pragma unroll
      for (int nf = 0; nf < 4; ++nf) {
        const float v = o[mf][nf][r] * inv;
        X2[rowX * 1024 + h * 64 + nf * 16 + colLane] = (__bf16)v;
      }
    }
}

// ---------------------------------------------------------------------------
// Kernel 3: output projection  (4096x1024 bf16)*(1024x1024) + bias -> fp32
// A tile is a pure bf16 copy -> async DMA to LDS; B tile converts via regs.
// ---------------------------------------------------------------------------
__global__ __launch_bounds__(256)
void proj_gemm_kernel(const __bf16* __restrict__ X2, const float* __restrict__ Wproj,
                      const float* __restrict__ bproj, float* __restrict__ out) {
  constexpr int AS = 40;
  constexpr int BS = 136;
  __shared__ __bf16 As[2][128 * AS];
  __shared__ __bf16 Bs[2][32 * BS];

  const int t = threadIdx.x;
  const int lane = t & 31;
  const int wave = t >> 5;
  const int mBase = blockIdx.y * 128;
  const int nBase = blockIdx.x * 128;
  const int waveM = (wave & 3) * 32;
  const int waveN = (wave >> 2) * 64;

  float4 bReg[4];
  const int ac = (t & 3) * 8,  ar = t >> 2;   // A: col 0..24, row 0..63 (+64i)
  const int bc = (t & 31) * 4, br = t >> 5;   // B: col 0..124, row 0..7 (+8i)

  auto asyncA = [&](int kk, int buf) {
    #pragma unroll
    for (int i = 0; i < 2; ++i)
      async_copy_b128(&As[buf][(ar + 64 * i) * AS + ac],
                      X2 + (size_t)(mBase + ar + 64 * i) * 1024 + kk + ac);
  };
  auto loadB = [&](int kk) {
    #pragma unroll
    for (int i = 0; i < 4; ++i)
      bReg[i] = *(const float4*)(Wproj + (size_t)(kk + br + 8 * i) * 1024 + nBase + bc);
  };
  auto storeB = [&](int buf) {
    #pragma unroll
    for (int i = 0; i < 4; ++i) {
      __bf16* db = &Bs[buf][(br + 8 * i) * BS + bc];
      db[0] = (__bf16)bReg[i].x; db[1] = (__bf16)bReg[i].y;
      db[2] = (__bf16)bReg[i].z; db[3] = (__bf16)bReg[i].w;
    }
  };

  v8f acc[2][4];
  #pragma unroll
  for (int i = 0; i < 2; ++i)
    #pragma unroll
    for (int j = 0; j < 4; ++j) acc[i][j] = vzero8f();

  asyncA(0, 0);
  loadB(0);
  storeB(0);
  async_wait0();
  __syncthreads();

  int cur = 0;
  for (int k0 = 0; k0 < 1024; k0 += 32) {
    const bool more = (k0 + 32) < 1024;
    if (more) { asyncA(k0 + 32, cur ^ 1); loadB(k0 + 32); }
    if (k0 + 64 < 1024) {
      __builtin_prefetch(X2 + (size_t)(mBase + ar) * 1024 + k0 + 64 + ac, 0, 1);
      __builtin_prefetch(Wproj + (size_t)(k0 + 64 + br) * 1024 + nBase + bc, 0, 1);
    }

    const v16bf a0 = load_a_frag(&As[cur][(waveM +  0) * AS], AS);
    const v16bf a1 = load_a_frag(&As[cur][(waveM + 16) * AS], AS);
    v16bf bf[4];
    #pragma unroll
    for (int nf = 0; nf < 4; ++nf)
      bf[nf] = load_b_frag(&Bs[cur][waveN + nf * 16], BS);
    SCHED_FENCE();
    #pragma unroll
    for (int nf = 0; nf < 4; ++nf) {
      acc[0][nf] = wmma_bf16(a0, bf[nf], acc[0][nf]);
      acc[1][nf] = wmma_bf16(a1, bf[nf], acc[1][nf]);
    }
    SCHED_FENCE();

    if (more) {
      storeB(cur ^ 1);
      async_wait0();
      __syncthreads();
      cur ^= 1;
    }
  }

  const int colLane = lane & 15;
  const int rowSel = (lane >> 4) << 3;
  #pragma unroll
  for (int mf = 0; mf < 2; ++mf)
    #pragma unroll
    for (int nf = 0; nf < 4; ++nf) {
      const int col = nBase + waveN + nf * 16 + colLane;
      const float bias = bproj[col];
      #pragma unroll
      for (int r = 0; r < 8; ++r) {
        const int row = mBase + waveM + mf * 16 + r + rowSel;
        out[(size_t)row * 1024 + col] = acc[mf][nf][r] + bias;
      }
    }
}

// ---------------------------------------------------------------------------
extern "C" void kernel_launch(void* const* d_in, const int* in_sizes, int n_in,
                              void* d_out, int out_size, void* d_ws, size_t ws_size,
                              hipStream_t stream) {
  const float* x     = (const float*)d_in[0];
  const float* Wqkv  = (const float*)d_in[1];
  const float* bqkv  = (const float*)d_in[2];
  const float* Wproj = (const float*)d_in[3];
  const float* bproj = (const float*)d_in[4];
  float* out = (float*)d_out;

  // Workspace: 4 bf16 arrays of 4M elements each (8 MB each, 32 MB total).
  __bf16* Qs = (__bf16*)d_ws;
  __bf16* Kt = Qs + (size_t)4 * 1024 * 1024;
  __bf16* Vs = Kt + (size_t)4 * 1024 * 1024;
  __bf16* X2 = Vs + (size_t)4 * 1024 * 1024;

  qkv_gemm_kernel<<<dim3(24, 32), 256, 0, stream>>>(x, Wqkv, bqkv, Qs, Kt, Vs);
  attn_kernel<<<dim3(8, 32), 256, 0, stream>>>(Qs, Kt, Vs, X2);
  proj_gemm_kernel<<<dim3(8, 32), 256, 0, stream>>>(X2, Wproj, bproj, out);
}
